// AttentionBlock_66511863546397
// MI455X (gfx1250) — matrix operand order repaired and checked by reading, not compiled, not run on hardware
//
#include <hip/hip_runtime.h>
#include <hip/hip_bf16.h>

typedef __attribute__((ext_vector_type(16))) _Float16 v16h;
typedef __attribute__((ext_vector_type(8)))  _Float16 v8h;
typedef __attribute__((ext_vector_type(8)))  float    v8f;

#define SCALE_QK 0.35355339059327373f  // 1/sqrt(sqrt(64))

// ---------------- WMMA fragment helpers (wave32, gfx1250 layouts) ----------------

// A-matrix 16x32 f16, row-major source with leading dim ld.
// Lane L: row m = row0 + (L&15); e0..7 -> K = k0 + (L>>4)*8 + e ; e8..15 -> K = k0 + 16 + (L>>4)*8 + (e-8)
__device__ __forceinline__ v16h load_a_rm(const _Float16* __restrict__ src, int row0, int k0, int ld) {
  int lane = threadIdx.x & 31;
  int m = row0 + (lane & 15);
  int h = lane >> 4;
  const _Float16* base = src + (size_t)m * ld + k0 + h * 8;
  v8h lo = *(const v8h*)(base);
  v8h hi = *(const v8h*)(base + 16);
  union { v16h v; v8h p[2]; } u;
  u.p[0] = lo; u.p[1] = hi;
  return u.v;
}

// B-matrix 32x16 f16, source stored column-contiguous: element (k, n) at src[n*ld + k].
// Lane L: col n = col0 + (L&15); e0..7 -> K = k0 + (L>>4)*8 + e ; e8..15 -> K = k0 + 16 + (L>>4)*8 + (e-8)
__device__ __forceinline__ v16h load_b_cm(const _Float16* src, int k0, int col0, int ld) {
  int lane = threadIdx.x & 31;
  int n = col0 + (lane & 15);
  int h = lane >> 4;
  const _Float16* base = src + (size_t)n * ld + k0 + h * 8;
  union { v16h v; v8h p[2]; } u;
  u.p[0] = *(const v8h*)(base);
  u.p[1] = *(const v8h*)(base + 16);
  return u.v;
}

__device__ __forceinline__ v8f wmma16(v16h a, v16h b, v8f c) {
  return __builtin_amdgcn_wmma_f32_16x16x32_f16(false, a, false, b, (short)0, c, false, false);
}

// ---------------- CDNA5 async global->LDS copy (ASYNCcnt-tracked) ----------------

__device__ __forceinline__ void async_copy_b128(unsigned int lds_off, const _Float16* g) {
  unsigned long long ga = (unsigned long long)g;
  asm volatile("global_load_async_to_lds_b128 %0, %1, off"
               :: "v"(lds_off), "v"(ga) : "memory");
}

// ---------------- GroupNorm ----------------

__global__ void gn_stats_kernel(const float* __restrict__ x, float* __restrict__ stats) {
  __shared__ float ssum[256];
  __shared__ float ssq[256];
  int bg = blockIdx.x;                       // (b*32 + g), 64 groups total
  const float* p = x + (size_t)bg * 65536;   // 16 channels * 4096 spatial, contiguous
  float s = 0.f, q = 0.f;
  for (int i = threadIdx.x; i < 65536; i += 256) {
    float v = p[i];
    s += v; q += v * v;
  }
  ssum[threadIdx.x] = s; ssq[threadIdx.x] = q;
  __syncthreads();
  for (int st = 128; st > 0; st >>= 1) {
    if ((int)threadIdx.x < st) {
      ssum[threadIdx.x] += ssum[threadIdx.x + st];
      ssq[threadIdx.x]  += ssq[threadIdx.x + st];
    }
    __syncthreads();
  }
  if (threadIdx.x == 0) {
    float mean = ssum[0] * (1.0f / 65536.0f);
    float var  = ssq[0] * (1.0f / 65536.0f) - mean * mean;
    stats[2 * bg]     = mean;
    stats[2 * bg + 1] = rsqrtf(var + 1e-6f);
  }
}

// Writes f32 normalized x into d_out (residual base) and f16 n-major copy xh[b][n][c].
__global__ void gn_apply_kernel(const float* __restrict__ x, const float* __restrict__ gw,
                                const float* __restrict__ gb, const float* __restrict__ stats,
                                float* __restrict__ outp, _Float16* __restrict__ xh) {
  size_t idx = (size_t)blockIdx.x * blockDim.x + threadIdx.x;  // exactly 2*512*4096 threads
  int n = (int)(idx & 4095);
  int c = (int)((idx >> 12) & 511);
  int b = (int)(idx >> 21);
  int bg = b * 32 + (c >> 4);
  float mu = stats[2 * bg], rs = stats[2 * bg + 1];
  float v = (x[idx] - mu) * rs * gw[c] + gb[c];
  outp[idx] = v;
  xh[((size_t)b * 4096 + n) * 512 + c] = (_Float16)v;
}

__global__ void f32_to_f16_kernel(const float* __restrict__ src, _Float16* __restrict__ dst, int n) {
  int i = blockIdx.x * blockDim.x + threadIdx.x;
  if (i < n) dst[i] = (_Float16)src[i];
}

// ---------------- QKV GEMM: C[1536,4096] = W[1536,512] * Xn[512,4096], per batch ----------------

__global__ void __launch_bounds__(256) qkv_gemm_kernel(const _Float16* __restrict__ wq,
                                                       const _Float16* __restrict__ xh,
                                                       const float* __restrict__ qkvb,
                                                       _Float16* __restrict__ qh,
                                                       _Float16* __restrict__ kh,
                                                       _Float16* __restrict__ vh) {
  int wid = blockIdx.x * 8 + (threadIdx.x >> 5);   // [0, 2*96*64)
  int b = wid / 6144;
  int rem = wid - b * 6144;
  int mt = rem >> 6;
  int ng = rem & 63;
  int m0 = mt * 16, n0 = ng * 64;
  const _Float16* xb = xh + (size_t)b * 4096 * 512;

  v8f cf[4] = {};
  for (int kc = 0; kc < 512; kc += 32) {
    v16h a = load_a_rm(wq, m0, kc, 512);
#pragma unroll
    for (int nt = 0; nt < 4; ++nt) {
      v16h bf = load_b_cm(xb, kc, n0 + nt * 16, 512);
      cf[nt] = wmma16(a, bf, cf[nt]);
    }
  }

  int lane = threadIdx.x & 31;
  int colb = lane & 15, hlf = lane >> 4;
#pragma unroll
  for (int nt = 0; nt < 4; ++nt) {
    int n = n0 + nt * 16 + colb;
#pragma unroll
    for (int r = 0; r < 8; ++r) {
      int orow = m0 + hlf * 8 + r;
      float val = cf[nt][r] + qkvb[orow];
      int head = orow / 192;
      int r3 = orow - head * 192;
      int Bh = b * 8 + head;
      if (r3 < 64)
        qh[((size_t)Bh * 4096 + n) * 64 + r3] = (_Float16)(val * SCALE_QK);
      else if (r3 < 128)
        kh[((size_t)Bh * 4096 + n) * 64 + (r3 - 64)] = (_Float16)(val * SCALE_QK);
      else
        vh[((size_t)Bh * 64 + (r3 - 128)) * 4096 + n] = (_Float16)val;
    }
  }
}

// ---------------- Flash attention ----------------
// Block = 8 waves = 256 threads, all on the same head; 8 x 16 = 128 queries per block.
// K (32x64) and V (64x32) tiles are staged into LDS with async copies (double buffered,
// each thread moves one 16B chunk -> one async instr per wave per tile), shared by all waves.

__global__ void __launch_bounds__(256) attn_kernel(const _Float16* __restrict__ qh,
                                                   const _Float16* __restrict__ kh,
                                                   const _Float16* __restrict__ vh,
                                                   _Float16* __restrict__ ah) {
  __shared__ __align__(32) _Float16 kbuf[2][32 * 64];  // [j_local][dd], ld=64, 4KB each
  __shared__ __align__(32) _Float16 vbuf[2][64 * 32];  // [dd][j_local], ld=32, 4KB each
  __shared__ __align__(32) _Float16 pbuf[8][16 * 32];  // per-wave P transpose tile

  int warp = threadIdx.x >> 5;
  int wid = blockIdx.x * 8 + warp;     // [0, 16*256)
  int Bh = wid >> 8;                   // head-batch index (0..15); same for all waves in block
  int i0 = (wid & 255) << 4;
  const _Float16* qb = qh + (size_t)Bh * 4096 * 64;
  const _Float16* kb = kh + (size_t)Bh * 4096 * 64;
  const _Float16* vb = vh + (size_t)Bh * 64 * 4096;
  int lane = threadIdx.x & 31;
  int colb = lane & 15, hlf = lane >> 4;
  _Float16* sh = pbuf[warp];

  unsigned int kof[2] = { (unsigned int)(size_t)&kbuf[0][0], (unsigned int)(size_t)&kbuf[1][0] };
  unsigned int vof[2] = { (unsigned int)(size_t)&vbuf[0][0], (unsigned int)(size_t)&vbuf[1][0] };

  v16h aq0 = load_a_rm(qb, i0, 0, 64);
  v16h aq1 = load_a_rm(qb, i0, 32, 64);
  v8f of[4] = {};
  float mrow[8], lrow[8];
#pragma unroll
  for (int r = 0; r < 8; ++r) { mrow[r] = -1e30f; lrow[r] = 0.f; }

  int t = threadIdx.x;                 // 0..255, moves 16B of K and 16B of V per fill
  int vdd = t >> 2;                    // V: 4 threads per dd row (64B)
  int vsub = (t & 3) * 8;              // element offset within V row chunk

  // prologue: fill buffer 0 with j0 = 0
  async_copy_b128(kof[0] + t * 16, kb + t * 8);
  async_copy_b128(vof[0] + t * 16, vb + (size_t)vdd * 4096 + vsub);

  int sel = 0;
  for (int j0 = 0; j0 < 4096; j0 += 32) {
    if (j0 + 32 < 4096) {
      // prefetch next tile into the other buffer, leave it in flight
      int jn = j0 + 32;
      async_copy_b128(kof[sel ^ 1] + t * 16, kb + (size_t)jn * 64 + t * 8);
      async_copy_b128(vof[sel ^ 1] + t * 16, vb + (size_t)vdd * 4096 + jn + vsub);
      asm volatile("s_wait_asynccnt 0x2" ::: "memory");  // current tile's 2 asyncs done
    } else {
      asm volatile("s_wait_asynccnt 0x0" ::: "memory");
    }
    __syncthreads();                                     // all waves' chunks visible

    const _Float16* kt = &kbuf[sel][0];
    const _Float16* vt = &vbuf[sel][0];

    // S tile 16 x 32 = two 16x16 C-fragments, K=64 over head dim (operands from LDS)
    v8f s0 = {}, s1 = {};
    v16h bk;
    bk = load_b_cm(kt, 0,  0,  64); s0 = wmma16(aq0, bk, s0);
    bk = load_b_cm(kt, 32, 0,  64); s0 = wmma16(aq1, bk, s0);
    bk = load_b_cm(kt, 0,  16, 64); s1 = wmma16(aq0, bk, s1);
    bk = load_b_cm(kt, 32, 16, 64); s1 = wmma16(aq1, bk, s1);

    // online softmax: row M = hlf*8 + r lives in the 16 lanes of this half
    float tmx[8];
#pragma unroll
    for (int r = 0; r < 8; ++r) tmx[r] = fmaxf(s0[r], s1[r]);
#pragma unroll
    for (int mask = 1; mask <= 8; mask <<= 1)
#pragma unroll
      for (int r = 0; r < 8; ++r) tmx[r] = fmaxf(tmx[r], __shfl_xor(tmx[r], mask, 32));

    float p0[8], p1[8], u[8], corr[8];
#pragma unroll
    for (int r = 0; r < 8; ++r) {
      float mn = fmaxf(mrow[r], tmx[r]);
      corr[r] = __expf(mrow[r] - mn);
      mrow[r] = mn;
      p0[r] = __expf(s0[r] - mn);
      p1[r] = __expf(s1[r] - mn);
      u[r] = p0[r] + p1[r];
    }
#pragma unroll
    for (int mask = 1; mask <= 8; mask <<= 1)
#pragma unroll
      for (int r = 0; r < 8; ++r) u[r] += __shfl_xor(u[r], mask, 32);
#pragma unroll
    for (int r = 0; r < 8; ++r) lrow[r] = lrow[r] * corr[r] + u[r];
#pragma unroll
    for (int dt = 0; dt < 4; ++dt)
#pragma unroll
      for (int r = 0; r < 8; ++r) of[dt][r] *= corr[r];

    // C-layout -> A-layout transpose of P through wave-private LDS (DS in-order per wave)
#pragma unroll
    for (int r = 0; r < 8; ++r) {
      int row = hlf * 8 + r;
      sh[row * 32 + colb]      = (_Float16)p0[r];
      sh[row * 32 + 16 + colb] = (_Float16)p1[r];
    }
    v8h lo = *(const v8h*)(sh + colb * 32 + hlf * 8);
    v8h hi = *(const v8h*)(sh + colb * 32 + 16 + hlf * 8);
    union { v16h v; v8h p[2]; } up;
    up.p[0] = lo; up.p[1] = hi;
    v16h pa = up.v;

    // out(16i x 64d) += P(16x32) * V^T(32x16) per 16-wide d tile (V from LDS)
#pragma unroll
    for (int dt = 0; dt < 4; ++dt) {
      v16h bv = load_b_cm(vt, 0, dt * 16, 32);
      of[dt] = wmma16(pa, bv, of[dt]);
    }

    __syncthreads();   // protect buffer from next iteration's prefetch overwrite
    sel ^= 1;
  }

  // finalize: divide by l, store a as n-major f16: ah[b][i][c], c = head*64 + dd
  int b = Bh >> 3, hd = Bh & 7;
  float linv[8];
#pragma unroll
  for (int r = 0; r < 8; ++r) linv[r] = 1.0f / lrow[r];
#pragma unroll
  for (int dt = 0; dt < 4; ++dt)
#pragma unroll
    for (int r = 0; r < 8; ++r) {
      int i = i0 + hlf * 8 + r;
      int cc = hd * 64 + dt * 16 + colb;
      ah[((size_t)b * 4096 + i) * 512 + cc] = (_Float16)(of[dt][r] * linv[r]);
    }
}

// ---------------- Projection GEMM: d_out[b][o][n] += Wp[o][c]*A[b][c][n] + bias ----------------

__global__ void __launch_bounds__(256) proj_gemm_kernel(const _Float16* __restrict__ wp,
                                                        const _Float16* __restrict__ ah,
                                                        const float* __restrict__ pb,
                                                        float* __restrict__ outp) {
  int wid = blockIdx.x * 8 + (threadIdx.x >> 5);   // [0, 2*32*64)
  int b = wid >> 11;
  int rem = wid & 2047;
  int mt = rem >> 6;
  int ng = rem & 63;
  int m0 = mt * 16, n0 = ng * 64;
  const _Float16* ab = ah + (size_t)b * 4096 * 512;

  v8f cf[4] = {};
  for (int kc = 0; kc < 512; kc += 32) {
    v16h a = load_a_rm(wp, m0, kc, 512);
#pragma unroll
    for (int nt = 0; nt < 4; ++nt) {
      v16h bf = load_b_cm(ab, kc, n0 + nt * 16, 512);
      cf[nt] = wmma16(a, bf, cf[nt]);
    }
  }

  int lane = threadIdx.x & 31;
  int colb = lane & 15, hlf = lane >> 4;
#pragma unroll
  for (int nt = 0; nt < 4; ++nt) {
    int n = n0 + nt * 16 + colb;
#pragma unroll
    for (int r = 0; r < 8; ++r) {
      int o = m0 + hlf * 8 + r;
      size_t idx = ((size_t)b * 512 + o) * 4096 + n;
      outp[idx] += cf[nt][r] + pb[o];
    }
  }
}

// ---------------- Host launch ----------------

extern "C" void kernel_launch(void* const* d_in, const int* in_sizes, int n_in,
                              void* d_out, int out_size, void* d_ws, size_t ws_size,
                              hipStream_t stream) {
  (void)in_sizes; (void)n_in; (void)out_size; (void)ws_size;
  const float* x     = (const float*)d_in[0];
  const float* gw    = (const float*)d_in[1];
  const float* gb    = (const float*)d_in[2];
  const float* qkvw  = (const float*)d_in[3];
  const float* qkvb  = (const float*)d_in[4];
  const float* projw = (const float*)d_in[5];
  const float* projb = (const float*)d_in[6];
  float* outp = (float*)d_out;

  char* ws = (char*)d_ws;
  float*    stats = (float*)(ws + 0);             //   512 B
  _Float16* wq16  = (_Float16*)(ws + 1024);       // 1536*512*2  = 1,572,864
  _Float16* wp16  = (_Float16*)(ws + 1573888);    //  512*512*2  =   524,288
  _Float16* xh    = (_Float16*)(ws + 2098176);    // 2*4096*512*2 = 8,388,608
  _Float16* qh    = (_Float16*)(ws + 10486784);   // 16*4096*64*2 = 8,388,608
  _Float16* kh    = (_Float16*)(ws + 18875392);   // 8,388,608
  _Float16* vh    = (_Float16*)(ws + 27264000);   // 8,388,608
  _Float16* ah    = (_Float16*)(ws + 35652608);   // 8,388,608  (total ~42 MB)

  f32_to_f16_kernel<<<(1536 * 512 + 255) / 256, 256, 0, stream>>>(qkvw, wq16, 1536 * 512);
  f32_to_f16_kernel<<<(512 * 512 + 255) / 256, 256, 0, stream>>>(projw, wp16, 512 * 512);
  gn_stats_kernel<<<64, 256, 0, stream>>>(x, stats);
  gn_apply_kernel<<<16384, 256, 0, stream>>>(x, gw, gb, stats, outp, xh);
  qkv_gemm_kernel<<<1536, 256, 0, stream>>>(wq16, xh, qkvb, qh, kh, vh);
  attn_kernel<<<512, 256, 0, stream>>>(qh, kh, vh, ah);
  proj_gemm_kernel<<<512, 256, 0, stream>>>(wp16, ah, projb, outp);
}
